// GroupContrastNet_87531433492890
// MI455X (gfx1250) — compile-verified, matrix operand order checked by reading, not run on hardware
//
#include <hip/hip_runtime.h>
#include <stdint.h>

typedef _Float16 v16h __attribute__((ext_vector_type(16)));
typedef _Float16 h8   __attribute__((ext_vector_type(8)));
typedef float    v8f  __attribute__((ext_vector_type(8)));

#define B_  2048
#define R_  256
#define D_  256
#define G_  32
#define S_  8

// CDNA5 hardware tanh (TRANS op, co-executes with VALU)
#if __has_builtin(__builtin_amdgcn_tanhf)
__device__ __forceinline__ float htanh(float x) { return __builtin_amdgcn_tanhf(x); }
#else
__device__ __forceinline__ float htanh(float x) {
    float r;
    asm("v_tanh_f32 %0, %1" : "=v"(r) : "v"(x));
    return r;
}
#endif

__device__ __forceinline__ float sigm(float v) {
    return __builtin_amdgcn_rcpf(1.0f + __expf(-v));
}

// ---------------- kernel 1: per-rule softmax over G, wdiff = wa - wb ----------------
__global__ void softmax_kernel(const float* __restrict__ selA,
                               const float* __restrict__ selB,
                               float* __restrict__ wdiff) {
    const int wave = threadIdx.x >> 5;
    const int lane = threadIdx.x & 31;
    const int r = blockIdx.x * 8 + wave;           // 32 blocks * 8 waves = 256 rules
    const int g = lane;

    float a = selA[r * G_ + g];
    float m = a;
    for (int o = 16; o > 0; o >>= 1) m = fmaxf(m, __shfl_xor(m, o, 32));
    float ea = __expf(a - m);
    float sa = ea;
    for (int o = 16; o > 0; o >>= 1) sa += __shfl_xor(sa, o, 32);

    float b = selB[r * G_ + g];
    float mb = b;
    for (int o = 16; o > 0; o >>= 1) mb = fmaxf(mb, __shfl_xor(mb, o, 32));
    float eb = __expf(b - mb);
    float sb = eb;
    for (int o = 16; o > 0; o >>= 1) sb += __shfl_xor(sb, o, 32);

    wdiff[r * G_ + g] = ea / sa - eb / sb;
}

// ---------------- kernel 2: d -> group map, head_w to f16 ----------------
__global__ void misc_kernel(const int* __restrict__ group_idx,
                            const float* __restrict__ head_w,
                            int* __restrict__ dmap,
                            _Float16* __restrict__ wh) {
    const int i = threadIdx.x;                     // 256 threads, 1 block
    dmap[group_idx[i]] = i >> 3;                   // flat (g,s) -> g  (groups partition D)
    wh[i] = (_Float16)head_w[i];
}

// ---------------- kernel 3: fold params into packed tiled tables ----------------
// 2*sigmoid(kappa*tanh(ineq)*(x-th)) - 1 == tanh(0.5*kappa*tanh(ineq)*(x-th))
// so k' = 0.5*kappa*tanh(ineq), b0' = -k'*th, A = sigm(mask)*tanh(esign)*wdiff
// layout: arr[rc=r/32][d4=d/4][rl=r%32][dj=d%4]  (8192 floats / 32KB per r-chunk)
__global__ void pack_kernel(const float* __restrict__ th,
                            const float* __restrict__ ineq,
                            const float* __restrict__ esign,
                            const float* __restrict__ mask,
                            const float* __restrict__ log_kappa,
                            const float* __restrict__ wdiff,
                            const int* __restrict__ dmap,
                            float* __restrict__ kpk,
                            float* __restrict__ bpk,
                            float* __restrict__ apk) {
    const int r = blockIdx.x;      // 256 blocks
    const int d = threadIdx.x;     // 256 threads
    const int idx = r * D_ + d;

    float kappa = fminf(fmaxf(__expf(log_kappa[0]), 0.5f), 50.0f);
    float k  = 0.5f * kappa * tanhf(ineq[idx]);
    float b0 = -k * th[idx];
    float aw = sigm(mask[idx]) * tanhf(esign[idx]) * wdiff[r * G_ + dmap[d]];

    const int rc = r >> 5, rl = r & 31, d4 = d >> 2, dj = d & 3;
    const int p = rc * 8192 + d4 * 128 + rl * 4 + dj;
    kpk[p] = k; bpk[p] = b0; apk[p] = aw;
}

// ---------------- CDNA5 async global->LDS copy helpers ----------------
__device__ __forceinline__ void async_cp16(void* lds, const void* g) {
    uint32_t lo = (uint32_t)(uintptr_t)lds;        // low 32 bits of LDS aperture addr = byte offset
    asm volatile("global_load_async_to_lds_b128 %0, %1, off"
                 :: "v"(lo), "v"(g) : "memory");
}
__device__ __forceinline__ void async_wait0() {
    asm volatile("s_wait_asynccnt 0" ::: "memory");
}

// ---------------- kernel 4: fused main pass -> z (f16, B x R) ----------------
// grid: (B/8, R/32); block 256 = 8 waves. wave w owns batch row b, lane owns rule r.
__global__ void __launch_bounds__(256)
main_kernel(const float* __restrict__ x,
            const float* __restrict__ kpk,
            const float* __restrict__ bpk,
            const float* __restrict__ apk,
            const float* __restrict__ t,
            _Float16* __restrict__ zh) {
    __shared__ __align__(16) float x_lds[8 * D_];     // 8KB
    __shared__ __align__(16) float k_ch[4096];        // 16KB (one d-phase: d4 0..31)
    __shared__ __align__(16) float b_ch[4096];
    __shared__ __align__(16) float a_ch[4096];

    const int tid  = threadIdx.x;
    const int w    = tid >> 5;
    const int lane = tid & 31;
    const int bt   = blockIdx.x;     // 0..255
    const int rc   = blockIdx.y;     // 0..7
    const int b    = bt * 8 + w;
    const int r    = rc * 32 + lane;

    const float4* gx = (const float4*)(x + (size_t)bt * 8 * D_);
    const float4* gk = (const float4*)(kpk + rc * 8192);
    const float4* gb = (const float4*)(bpk + rc * 8192);
    const float4* ga = (const float4*)(apk + rc * 8192);
    float4* lx = (float4*)x_lds;
    float4* lk = (float4*)k_ch;
    float4* lb = (float4*)b_ch;
    float4* la = (float4*)a_ch;

    // ---- stage x tile + phase-0 params (async DMA to LDS) ----
    for (int i = tid; i < 512; i += 256) async_cp16(&lx[i], &gx[i]);
    for (int i = tid; i < 1024; i += 256) {
        async_cp16(&lk[i], &gk[i]);
        async_cp16(&lb[i], &gb[i]);
        async_cp16(&la[i], &ga[i]);
    }
    async_wait0();
    __syncthreads();

    float acc = 0.0f;

    // ---- phase 0: d = 0..127 ----
    #pragma unroll 8
    for (int d4 = 0; d4 < 32; ++d4) {
        const float4 xv = *(const float4*)&x_lds[w * D_ + d4 * 4];    // wave-uniform broadcast
        const float4 kv = *(const float4*)&k_ch[d4 * 128 + lane * 4]; // conflict-free b128
        const float4 bv = *(const float4*)&b_ch[d4 * 128 + lane * 4];
        const float4 av = *(const float4*)&a_ch[d4 * 128 + lane * 4];
        acc = fmaf(av.x, htanh(fmaf(kv.x, xv.x, bv.x)), acc);
        acc = fmaf(av.y, htanh(fmaf(kv.y, xv.y, bv.y)), acc);
        acc = fmaf(av.z, htanh(fmaf(kv.z, xv.z, bv.z)), acc);
        acc = fmaf(av.w, htanh(fmaf(kv.w, xv.w, bv.w)), acc);
    }
    __syncthreads();

    // ---- stage phase-1 params ----
    for (int i = tid; i < 1024; i += 256) {
        async_cp16(&lk[i], &gk[1024 + i]);
        async_cp16(&lb[i], &gb[1024 + i]);
        async_cp16(&la[i], &ga[1024 + i]);
    }
    async_wait0();
    __syncthreads();

    // ---- phase 1: d = 128..255 ----
    #pragma unroll 8
    for (int d4 = 0; d4 < 32; ++d4) {
        const float4 xv = *(const float4*)&x_lds[w * D_ + (32 + d4) * 4];
        const float4 kv = *(const float4*)&k_ch[d4 * 128 + lane * 4];
        const float4 bv = *(const float4*)&b_ch[d4 * 128 + lane * 4];
        const float4 av = *(const float4*)&a_ch[d4 * 128 + lane * 4];
        acc = fmaf(av.x, htanh(fmaf(kv.x, xv.x, bv.x)), acc);
        acc = fmaf(av.y, htanh(fmaf(kv.y, xv.y, bv.y)), acc);
        acc = fmaf(av.z, htanh(fmaf(kv.z, xv.z, bv.z)), acc);
        acc = fmaf(av.w, htanh(fmaf(kv.w, xv.w, bv.w)), acc);
    }

    // z = sigmoid(8*(Ea-Eb-t)) = 0.5 + 0.5*tanh(4*(Ea-Eb-t))
    float zz = fmaf(0.5f, htanh(4.0f * (acc - t[r])), 0.5f);
    zh[(size_t)b * R_ + r] = (_Float16)zz;
}

// ---------------- kernel 5: head y = z @ w^T + b via WMMA f16 ----------------
// one wave per 16 batch rows; K = R = 256 in eight 16x16x32 WMMA steps.
__global__ void __launch_bounds__(32)
head_kernel(const _Float16* __restrict__ zh,
            const _Float16* __restrict__ wh,
            const float* __restrict__ head_b,
            float* __restrict__ y) {
    const int lane  = threadIdx.x;
    const int l16   = lane & 15;
    const int khalf = lane >> 4;      // 0: K 0..7/16..23 ; 1: K 8..15/24..31
    const int kbase = khalf * 8;
    const int tile  = blockIdx.x;     // 128 tiles of 16 rows
    const int brow  = tile * 16 + l16;

    v8f c = {};
    #pragma unroll
    for (int kk = 0; kk < 8; ++kk) {
        // A fragment: rows of z
        const _Float16* zr = zh + (size_t)brow * R_ + kk * 32;
        h8 alo = *(const h8*)(zr + kbase);
        h8 ahi = *(const h8*)(zr + kbase + 16);
        v16h a;
        #pragma unroll
        for (int j = 0; j < 8; ++j) { a[j] = alo[j]; a[j + 8] = ahi[j]; }

        // B fragment: column 0 = head_w slice, other columns 0
        v16h bw = {};
        if (l16 == 0) {
            h8 blo = *(const h8*)(wh + kk * 32 + kbase);
            h8 bhi = *(const h8*)(wh + kk * 32 + kbase + 16);
            #pragma unroll
            for (int j = 0; j < 8; ++j) { bw[j] = blo[j]; bw[j + 8] = bhi[j]; }
        }

        c = __builtin_amdgcn_wmma_f32_16x16x32_f16(
                false, a, false, bw, (short)0, c, false, false);
    }

    if (l16 == 0) {                   // column N==0 holds y
        const float hb = head_b[0];
        float* yo = y + tile * 16 + khalf * 8;
        float4 o0, o1;
        o0.x = c[0] + hb; o0.y = c[1] + hb; o0.z = c[2] + hb; o0.w = c[3] + hb;
        o1.x = c[4] + hb; o1.y = c[5] + hb; o1.z = c[6] + hb; o1.w = c[7] + hb;
        *(float4*)yo       = o0;
        *(float4*)(yo + 4) = o1;
    }
}

// ---------------- host launcher ----------------
extern "C" void kernel_launch(void* const* d_in, const int* in_sizes, int n_in,
                              void* d_out, int out_size, void* d_ws, size_t ws_size,
                              hipStream_t stream) {
    const float* x         = (const float*)d_in[0];
    const float* th        = (const float*)d_in[1];
    const float* ineq      = (const float*)d_in[2];
    const float* esign     = (const float*)d_in[3];
    const float* mask      = (const float*)d_in[4];
    const float* log_kappa = (const float*)d_in[5];
    const float* selA      = (const float*)d_in[6];
    const float* selB      = (const float*)d_in[7];
    const float* t         = (const float*)d_in[8];
    const float* head_w    = (const float*)d_in[9];
    const float* head_b    = (const float*)d_in[10];
    const int*   group_idx = (const int*)d_in[11];
    float* y = (float*)d_out;

    char* w8 = (char*)d_ws;
    float*     wdiff = (float*)(w8);                    //  32 KB
    int*       dmap  = (int*)(w8 + (32u << 10));        //   1 KB
    _Float16*  wh    = (_Float16*)(w8 + (36u << 10));   // 0.5 KB
    float*     kpk   = (float*)(w8 + (64u << 10));      // 256 KB
    float*     bpk   = (float*)(w8 + (320u << 10));     // 256 KB
    float*     apk   = (float*)(w8 + (576u << 10));     // 256 KB
    _Float16*  zh    = (_Float16*)(w8 + (832u << 10));  //   1 MB

    softmax_kernel<<<R_ / 8, 256, 0, stream>>>(selA, selB, wdiff);
    misc_kernel<<<1, 256, 0, stream>>>(group_idx, head_w, dmap, wh);
    pack_kernel<<<R_, 256, 0, stream>>>(th, ineq, esign, mask, log_kappa,
                                        wdiff, dmap, kpk, bpk, apk);
    main_kernel<<<dim3(B_ / 8, R_ / 32), 256, 0, stream>>>(x, kpk, bpk, apk, t, zh);
    head_kernel<<<B_ / 16, 32, 0, stream>>>(zh, wh, head_b, y);
}
